// MoELayer_52673478918819
// MI455X (gfx1250) — matrix operand order repaired and checked by reading, not compiled, not run on hardware
//
#include <hip/hip_runtime.h>
#include <hip/hip_bf16.h>
#include <math.h>

typedef __attribute__((ext_vector_type(16))) __bf16 v16bf;
typedef __attribute__((ext_vector_type(8)))  float  v8f;

static constexpr int T   = 4096;   // B*S tokens
static constexpr int DM  = 1024;   // d_model
static constexpr int DF  = 4096;   // d_ff
static constexpr int NE  = 24;     // experts
static constexpr int TM  = 128;    // GEMM tile M (tokens)
static constexpr int TN  = 128;    // GEMM tile N
static constexpr int TK  = 32;     // GEMM tile K (bf16 wmma K)
static constexpr int LDP = 40;     // padded LDS row stride in halves (80B, 16B-aligned)

union FragBF { v16bf v; uint4 u4[2]; };

// Hardware bf16 convert (RNE on gfx1250 VALU).
__device__ __forceinline__ unsigned short f2bf(float f) {
  union { __bf16 h; unsigned short u; } c;
  c.h = (__bf16)f;
  return c.u;
}

__device__ __forceinline__ float gelu_exact(float v) {
  return 0.5f * v * (1.0f + erff(v * 0.70710678118654752440f));
}

// Async global->LDS copy of 32 contiguous bytes (two b128), tracked by ASYNCcnt.
// IOFFSET applies to both the LDS and global addresses, so one (lds, gaddr)
// pair covers both halves.
__device__ __forceinline__ void async_copy_b256(unsigned short* lds_dst,
                                                const unsigned short* gsrc) {
  unsigned lds = (unsigned)(unsigned long long)lds_dst;     // low 32 bits = LDS offset
  unsigned long long ga = (unsigned long long)gsrc;
  asm volatile(
      "global_load_async_to_lds_b128 %0, %1, off\n\t"
      "global_load_async_to_lds_b128 %0, %1, off offset:16"
      :: "v"(lds), "v"(ga) : "memory");
}

__device__ __forceinline__ void wait_async_zero() {
  asm volatile("s_wait_asynccnt 0x0" ::: "memory");
}

// ---------------------------------------------------------------- convert x -> bf16
__global__ void k_convert(const float* __restrict__ x, unsigned short* __restrict__ xb) {
  size_t i = ((size_t)blockIdx.x * blockDim.x + threadIdx.x) * 4;
  float4 f = *(const float4*)(x + i);
  uint2 p;
  p.x = (unsigned)f2bf(f.x) | ((unsigned)f2bf(f.y) << 16);
  p.y = (unsigned)f2bf(f.z) | ((unsigned)f2bf(f.w) << 16);
  *(uint2*)(xb + i) = p;
}

// ---------------------------------------------------------------- gating + routing
__global__ void k_gate_route(const float* __restrict__ x, const float* __restrict__ gw,
                             const float* __restrict__ gb, int* __restrict__ counts,
                             int* __restrict__ tok, int* __restrict__ slt,
                             float* __restrict__ wgt) {
  int t = blockIdx.x;
  __shared__ float sc[NE];
  int tid = threadIdx.x;          // 128 threads = 4 waves
  int wave = tid >> 5, lane = tid & 31;
  const float* xr = x + (size_t)t * DM;
  #pragma unroll
  for (int ei = 0; ei < NE / 4; ++ei) {
    int e = wave * (NE / 4) + ei;
    float s = 0.f;
    for (int d = lane; d < DM; d += 32) s += xr[d] * gw[(size_t)d * NE + e];
    #pragma unroll
    for (int off = 16; off > 0; off >>= 1) s += __shfl_down(s, off, 32);
    if (lane == 0) sc[e] = s + gb[e];
  }
  __syncthreads();
  if (tid == 0) {
    int i0 = -1, i1 = -1; float v0 = -3.4e38f, v1 = -3.4e38f;
    #pragma unroll
    for (int e = 0; e < NE; ++e) {
      float v = sc[e];
      if (v > v0) { v1 = v0; i1 = i0; v0 = v; i0 = e; }
      else if (v > v1) { v1 = v; i1 = e; }
    }
    float e1 = __expf(v1 - v0);
    float inv = 1.f / (1.f + e1);
    int p0 = atomicAdd(&counts[i0], 1);
    tok[i0 * T + p0] = t; slt[i0 * T + p0] = 2 * t;     wgt[i0 * T + p0] = inv;
    int p1 = atomicAdd(&counts[i1], 1);
    tok[i1 * T + p1] = t; slt[i1 * T + p1] = 2 * t + 1; wgt[i1 * T + p1] = e1 * inv;
  }
}

// ---------------------------------------------------------------- expert FFN layer 1
// h[slot] = gelu( gather(xb) @ w1[e] + b1[e] ),  h stored bf16, [2T, DF]
__global__ void k_ffn1(const unsigned short* __restrict__ xb, const float* __restrict__ w1,
                       const float* __restrict__ b1, const int* __restrict__ counts,
                       const int* __restrict__ tok, const int* __restrict__ slt,
                       unsigned short* __restrict__ hbuf) {
  int e = blockIdx.z;
  int ne = counts[e];
  int row0 = blockIdx.y * TM;
  if (row0 >= ne) return;
  int nrows = min(TM, ne - row0);
  int n0 = blockIdx.x * TN;

  __shared__ unsigned short As[TM][LDP];
  __shared__ unsigned short Bs[TN][LDP];
  __shared__ int srow[TM];
  __shared__ int sslot[TM];

  int tid = threadIdx.x;          // 256 threads = 8 waves
  if (tid < TM) {
    int ok = tid < nrows;
    srow[tid]  = ok ? tok[e * T + row0 + tid] : 0;
    sslot[tid] = ok ? slt[e * T + row0 + tid] : 0;
  }
  __syncthreads();

  int wave = tid >> 5, lane = tid & 31;
  int wm = wave & 3, wn = wave >> 2;        // wave tile: 32 rows x 64 cols
  v8f acc[2][4] = {};

  int lr  = lane & 15;
  int kbA = (lane < 16) ? 0 : 8;            // A: 16x32 bf16 layout
  int kbB = (lane < 16) ? 0 : 8;            // B: 32x16 bf16 layout

  int ar = tid >> 1, ac = (tid & 1) * 16;   // A-stage: one 32B half-row per thread
  const unsigned short* asrc = xb + (size_t)srow[ar] * DM + ac;

  for (int k0 = 0; k0 < DM; k0 += TK) {
    { // stage A: async DMA global->LDS (bf16 gathered rows), zero-fill padding
      if (ar < nrows) {
        async_copy_b256(&As[ar][ac], asrc + k0);
      } else {
        uint4 z = make_uint4(0u, 0u, 0u, 0u);
        *(uint4*)&As[ar][ac]     = z;
        *(uint4*)&As[ar][ac + 8] = z;
      }
    }
    { // stage B: w1[e][k0..k0+31][n0..n0+127] fp32 -> bf16, transposed [n][k]
      int kk = tid >> 3, nb = (tid & 7) * 16;
      const float4* src = (const float4*)(w1 + (size_t)e * DM * DF +
                                          (size_t)(k0 + kk) * DF + n0 + nb);
      __builtin_prefetch((const char*)src + (size_t)TK * DF * sizeof(float), 0, 1);
      #pragma unroll
      for (int j = 0; j < 4; ++j) {
        float4 f = src[j];
        Bs[nb + 4 * j + 0][kk] = f2bf(f.x);
        Bs[nb + 4 * j + 1][kk] = f2bf(f.y);
        Bs[nb + 4 * j + 2][kk] = f2bf(f.z);
        Bs[nb + 4 * j + 3][kk] = f2bf(f.w);
      }
    }
    wait_async_zero();
    __syncthreads();

    FragBF a[2];
    #pragma unroll
    for (int am = 0; am < 2; ++am) {
      int row = 32 * wm + 16 * am + lr;
      a[am].u4[0] = *(const uint4*)&As[row][kbA];
      a[am].u4[1] = *(const uint4*)&As[row][kbA + 16];
    }
    #pragma unroll
    for (int ns = 0; ns < 4; ++ns) {
      FragBF b;
      int col = 64 * wn + 16 * ns + lr;
      b.u4[0] = *(const uint4*)&Bs[col][kbB];
      b.u4[1] = *(const uint4*)&Bs[col][kbB + 16];
      #pragma unroll
      for (int am = 0; am < 2; ++am)
        acc[am][ns] = __builtin_amdgcn_wmma_f32_16x16x32_bf16(
            false, a[am].v, false, b.v, (short)0, acc[am][ns], false, false);
    }
    __syncthreads();
  }

  int mh = (lane >> 4) * 8;                 // C/D layout: M = r + 8*(lane>=16)
  #pragma unroll
  for (int ns = 0; ns < 4; ++ns) {
    int col = n0 + 64 * wn + 16 * ns + lr;
    float bb = b1[(size_t)e * DF + col];
    #pragma unroll
    for (int am = 0; am < 2; ++am) {
      #pragma unroll
      for (int r = 0; r < 8; ++r) {
        int m = 32 * wm + 16 * am + r + mh;
        if (m < nrows) {
          float v = gelu_exact(acc[am][ns][r] + bb);
          hbuf[(size_t)sslot[m] * DF + col] = f2bf(v);
        }
      }
    }
  }
}

// ---------------------------------------------------------------- expert FFN layer 2
// ybuf[slot] = gate * ( h[slot] @ w2[e] + b2[e] ),  fp32 [2T, DM]
__global__ void k_ffn2(const unsigned short* __restrict__ hbuf, const float* __restrict__ w2,
                       const float* __restrict__ b2, const int* __restrict__ counts,
                       const int* __restrict__ slt, const float* __restrict__ wgt,
                       float* __restrict__ ybuf) {
  int e = blockIdx.z;
  int ne = counts[e];
  int row0 = blockIdx.y * TM;
  if (row0 >= ne) return;
  int nrows = min(TM, ne - row0);
  int n0 = blockIdx.x * TN;

  __shared__ unsigned short As[TM][LDP];
  __shared__ unsigned short Bs[TN][LDP];
  __shared__ int   sslot[TM];
  __shared__ float swgt[TM];

  int tid = threadIdx.x;
  if (tid < TM) {
    int ok = tid < nrows;
    sslot[tid] = ok ? slt[e * T + row0 + tid] : 0;
    swgt[tid]  = ok ? wgt[e * T + row0 + tid] : 0.f;
  }
  __syncthreads();

  int wave = tid >> 5, lane = tid & 31;
  int wm = wave & 3, wn = wave >> 2;
  v8f acc[2][4] = {};

  int lr  = lane & 15;
  int kbA = (lane < 16) ? 0 : 8;
  int kbB = (lane < 16) ? 0 : 8;

  int ar = tid >> 1, ac = (tid & 1) * 16;
  const unsigned short* asrc = hbuf + (size_t)sslot[ar] * DF + ac;

  for (int k0 = 0; k0 < DF; k0 += TK) {
    { // stage A: async DMA global->LDS (bf16 h rows), zero-fill padding
      if (ar < nrows) {
        async_copy_b256(&As[ar][ac], asrc + k0);
      } else {
        uint4 z = make_uint4(0u, 0u, 0u, 0u);
        *(uint4*)&As[ar][ac]     = z;
        *(uint4*)&As[ar][ac + 8] = z;
      }
    }
    { // stage B: w2[e][k0..k0+31][n0..n0+127] fp32 -> bf16, transposed [n][k]
      int kk = tid >> 3, nb = (tid & 7) * 16;
      const float4* src = (const float4*)(w2 + (size_t)e * DF * DM +
                                          (size_t)(k0 + kk) * DM + n0 + nb);
      __builtin_prefetch((const char*)src + (size_t)TK * DM * sizeof(float), 0, 1);
      #pragma unroll
      for (int j = 0; j < 4; ++j) {
        float4 f = src[j];
        Bs[nb + 4 * j + 0][kk] = f2bf(f.x);
        Bs[nb + 4 * j + 1][kk] = f2bf(f.y);
        Bs[nb + 4 * j + 2][kk] = f2bf(f.z);
        Bs[nb + 4 * j + 3][kk] = f2bf(f.w);
      }
    }
    wait_async_zero();
    __syncthreads();

    FragBF a[2];
    #pragma unroll
    for (int am = 0; am < 2; ++am) {
      int row = 32 * wm + 16 * am + lr;
      a[am].u4[0] = *(const uint4*)&As[row][kbA];
      a[am].u4[1] = *(const uint4*)&As[row][kbA + 16];
    }
    #pragma unroll
    for (int ns = 0; ns < 4; ++ns) {
      FragBF b;
      int col = 64 * wn + 16 * ns + lr;
      b.u4[0] = *(const uint4*)&Bs[col][kbB];
      b.u4[1] = *(const uint4*)&Bs[col][kbB + 16];
      #pragma unroll
      for (int am = 0; am < 2; ++am)
        acc[am][ns] = __builtin_amdgcn_wmma_f32_16x16x32_bf16(
            false, a[am].v, false, b.v, (short)0, acc[am][ns], false, false);
    }
    __syncthreads();
  }

  int mh = (lane >> 4) * 8;
  #pragma unroll
  for (int ns = 0; ns < 4; ++ns) {
    int col = n0 + 64 * wn + 16 * ns + lr;
    float bb = b2[(size_t)e * DM + col];
    #pragma unroll
    for (int am = 0; am < 2; ++am) {
      #pragma unroll
      for (int r = 0; r < 8; ++r) {
        int m = 32 * wm + 16 * am + r + mh;
        if (m < nrows)
          ybuf[(size_t)sslot[m] * DM + col] = swgt[m] * (acc[am][ns][r] + bb);
      }
    }
  }
}

// ---------------------------------------------------------------- combine top-2
__global__ void k_combine(const float* __restrict__ ybuf, float* __restrict__ out) {
  size_t i4 = (size_t)blockIdx.x * blockDim.x + threadIdx.x;  // over T*DM/4
  size_t t  = i4 / (DM / 4);
  size_t d4 = i4 % (DM / 4);
  const float4* y = (const float4*)ybuf;
  float4 a = y[(2 * t) * (DM / 4) + d4];
  float4 b = y[(2 * t + 1) * (DM / 4) + d4];
  float4 o; o.x = a.x + b.x; o.y = a.y + b.y; o.z = a.z + b.z; o.w = a.w + b.w;
  ((float4*)out)[i4] = o;
}

// ---------------------------------------------------------------- launch
extern "C" void kernel_launch(void* const* d_in, const int* in_sizes, int n_in,
                              void* d_out, int out_size, void* d_ws, size_t ws_size,
                              hipStream_t stream) {
  (void)in_sizes; (void)n_in; (void)out_size; (void)ws_size;
  const float* x  = (const float*)d_in[0];
  const float* gw = (const float*)d_in[1];
  const float* gb = (const float*)d_in[2];
  const float* w1 = (const float*)d_in[3];
  const float* b1 = (const float*)d_in[4];
  const float* w2 = (const float*)d_in[5];
  const float* b2 = (const float*)d_in[6];
  float* out = (float*)d_out;

  char* ws = (char*)d_ws;
  size_t off = 0;
  unsigned short* xb = (unsigned short*)(ws + off); off += (size_t)T * DM * 2;   // 8 MB
  int*   counts = (int*)(ws + off);   off += 256;
  int*   tok    = (int*)(ws + off);   off += (size_t)NE * T * 4;                 // 384 KB
  int*   slt    = (int*)(ws + off);   off += (size_t)NE * T * 4;
  float* wgt    = (float*)(ws + off); off += (size_t)NE * T * 4;
  unsigned short* hbuf = (unsigned short*)(ws + off); off += (size_t)2 * T * DF * 2; // 64 MB
  float* ybuf   = (float*)(ws + off); off += (size_t)2 * T * DM * 4;             // 32 MB

  hipMemsetAsync(counts, 0, 256, stream);

  k_convert<<<(T * DM / 4) / 256, 256, 0, stream>>>(x, xb);
  k_gate_route<<<T, 128, 0, stream>>>(x, gw, gb, counts, tok, slt, wgt);
  k_ffn1<<<dim3(DF / TN, T / TM, NE), 256, 0, stream>>>(xb, w1, b1, counts, tok, slt, hbuf);
  k_ffn2<<<dim3(DM / TN, T / TM, NE), 256, 0, stream>>>(hbuf, w2, b2, counts, slt, wgt, ybuf);
  k_combine<<<(T * DM / 4) / 256, 256, 0, stream>>>(ybuf, out);
}
